// JiTAttnMixer_82317343195235
// MI455X (gfx1250) — matrix operand order repaired and checked by reading, not compiled, not run on hardware
//
#include <hip/hip_runtime.h>

#define DIM   768
#define DIM3  2304
#define NHEAD 12
#define HD    64
#define NB    4
#define SEQ   2048
#define MTOT  (NB * SEQ)   // 8192 tokens

typedef __attribute__((ext_vector_type(16))) __bf16 v16bf;
typedef __attribute__((ext_vector_type(8)))  __bf16 v8bf;
typedef __attribute__((ext_vector_type(8)))  float  v8f;

union V16U { v16bf v; v8bf h[2]; };

// fp32 -> bf16, round-to-nearest-even
static __device__ __forceinline__ __bf16 f2bf(float f) {
    unsigned u = __builtin_bit_cast(unsigned, f);
    u += 0x7FFFu + ((u >> 16) & 1u);
    unsigned short s = (unsigned short)(u >> 16);
    return __builtin_bit_cast(__bf16, s);
}

static __device__ __forceinline__ v8f wmma_bf16(v16bf a, v16bf b, v8f c) {
    // D = A(16x32 bf16) x B(32x16 bf16) + C(16x16 f32)
    return __builtin_amdgcn_wmma_f32_16x16x32_bf16(
        /*neg_a=*/false, a, /*neg_b=*/false, b,
        /*c_mod=*/(short)0, c, /*reuse_a=*/false, /*reuse_b=*/false);
}

// ---------------------------------------------------------------- converts
__global__ void k_cvt(const float* __restrict__ in, __bf16* __restrict__ out, int n) {
    for (int i = blockIdx.x * blockDim.x + threadIdx.x; i < n; i += gridDim.x * blockDim.x)
        out[i] = f2bf(in[i]);
}

// ------------------------------------------------- QKV GEMM + bias + RMS-norm
// X: [8192,768] bf16 ; W: [2304,768] bf16 (row-major; B-frag = row of W)
// wave tile: 32 rows x 64 cols (2 A-frags share every B-frag -> 8 WMMA / k-chunk)
// col-block j in [0,36): j<12 q-head, <24 k-head, else v-head
__global__ void __launch_bounds__(256) k_gemm_qkv(
    const __bf16* __restrict__ X, const __bf16* __restrict__ W,
    const float* __restrict__ bias, const float* __restrict__ qnw,
    const float* __restrict__ knw,
    __bf16* __restrict__ q_bf, __bf16* __restrict__ k_bf, __bf16* __restrict__ v_t)
{
    const int lane = threadIdx.x & 31;
    const int l = lane & 15, hf = lane >> 4;
    const int wave = threadIdx.x >> 5;
    const int m0 = (blockIdx.y * 8 + wave) * 32;
    const int j  = blockIdx.x;
    const int n0 = j * 64;

    v8f c[2][4] = {};
    const __bf16* pa0 = X + (size_t)(m0 + l) * DIM + hf * 16;        // A: lane=row
    const __bf16* pa1 = X + (size_t)(m0 + 16 + l) * DIM + hf * 16;
    for (int k0 = 0; k0 < DIM; k0 += 32) {
        V16U a0, a1;
        a0.h[0] = *(const v8bf*)(pa0 + k0);
        a0.h[1] = *(const v8bf*)(pa0 + k0 + 8);
        a1.h[0] = *(const v8bf*)(pa1 + k0);
        a1.h[1] = *(const v8bf*)(pa1 + k0 + 8);
#pragma unroll
        for (int n = 0; n < 4; ++n) {
            // B[k, col] = W[col, k]; lane=col, K contiguous per half
            v16bf b = *(const v16bf*)(W + (size_t)(n0 + n * 16 + l) * DIM + k0 + hf * 16);
            c[0][n] = wmma_bf16(a0.v, b, c[0][n]);
            c[1][n] = wmma_bf16(a1.v, b, c[1][n]);
        }
    }

    // bias (pre-norm, matches reference)
#pragma unroll
    for (int n = 0; n < 4; ++n) {
        const float bv = bias[n0 + n * 16 + l];
#pragma unroll
        for (int r = 0; r < 2; ++r)
#pragma unroll
            for (int i = 0; i < 8; ++i) c[r][n][i] += bv;
    }

    if (j < 24) {  // q or k head: RMS-norm over the 64 cols (exactly this wave tile)
        const float* nw = (j < 12) ? qnw : knw;
        const int head  = (j < 12) ? j : j - 12;
        __bf16* dst     = (j < 12) ? q_bf : k_bf;
        // fold attention scale 1/sqrt(64) into stored q
        const float post = (j < 12) ? 0.125f : 1.0f;
        float nwc[4];
#pragma unroll
        for (int n = 0; n < 4; ++n) nwc[n] = nw[n * 16 + l] * post;
#pragma unroll
        for (int r = 0; r < 2; ++r)
#pragma unroll
            for (int i = 0; i < 8; ++i) {
                float ss = 0.f;
#pragma unroll
                for (int n = 0; n < 4; ++n) ss += c[r][n][i] * c[r][n][i];
                ss += __shfl_xor(ss, 8, 32);
                ss += __shfl_xor(ss, 4, 32);
                ss += __shfl_xor(ss, 2, 32);
                ss += __shfl_xor(ss, 1, 32);
                const float rn = rsqrtf(ss * (1.0f / 64.0f) + 1e-6f);
                const int t = m0 + r * 16 + i + hf * 8;
                const int bb = t >> 11, ntok = t & 2047;
                const size_t base = ((size_t)(bb * NHEAD + head) * SEQ + ntok) * HD;
#pragma unroll
                for (int n = 0; n < 4; ++n)
                    dst[base + n * 16 + l] = f2bf(c[r][n][i] * rn * nwc[n]);
            }
    } else {       // v head: store transposed [b,h,d,n] for PV B-fragments
        const int head = j - 24;
#pragma unroll
        for (int r = 0; r < 2; ++r)
#pragma unroll
            for (int i = 0; i < 8; ++i) {
                const int t = m0 + r * 16 + i + hf * 8;
                const int bb = t >> 11, ntok = t & 2047;
                const size_t base = (size_t)(bb * NHEAD + head) * HD * SEQ;
#pragma unroll
                for (int n = 0; n < 4; ++n)
                    v_t[base + (size_t)(n * 16 + l) * SEQ + ntok] = f2bf(c[r][n][i]);
            }
    }
}

// ----------------------------------------------------------- flash attention
// one wave = one 16-row query tile, streaming 32-key tiles.
// q,k are RMS-normalized (|logit| <= 8 by Cauchy-Schwarz; scale folded into q),
// so softmax uses a FIXED shift exp(s-8): no overflow possible, no online max,
// no in-loop cross-lane reductions -- per-lane partial sums, one reduce at end.
__global__ void __launch_bounds__(128) k_flash(
    const __bf16* __restrict__ q_bf, const __bf16* __restrict__ k_bf,
    const __bf16* __restrict__ v_t, __bf16* __restrict__ attn)
{
    __shared__ __bf16 lds_p[4][16 * 32];   // per-wave P-tile transpose buffer
    const int lane = threadIdx.x & 31;
    const int l = lane & 15, hf = lane >> 4;
    const int wave = threadIdx.x >> 5;
    const int wg = blockIdx.x * 4 + wave;
    const int bh = wg >> 7;                // 48 (b,h) pairs x 128 q-tiles
    const int qt = (wg & 127) << 4;
    const int b = bh / NHEAD, h = bh % NHEAD;
    const __bf16* Q  = q_bf + (size_t)bh * SEQ * HD;
    const __bf16* K  = k_bf + (size_t)bh * SEQ * HD;
    const __bf16* Vt = v_t  + (size_t)bh * HD * SEQ;
    __bf16* myp = lds_p[wave];

    v16bf qa[2];
#pragma unroll
    for (int dc = 0; dc < 2; ++dc) {
        const __bf16* p = Q + (size_t)(qt + l) * HD + dc * 32 + hf * 16;
        V16U t; t.h[0] = *(const v8bf*)p; t.h[1] = *(const v8bf*)(p + 8);
        qa[dc] = t.v;
    }

    v8f o[4] = {};
    float lsum[8];
#pragma unroll
    for (int i = 0; i < 8; ++i) lsum[i] = 0.f;

    for (int kt = 0; kt < SEQ; kt += 32) {
        v8f s[2] = {};
#pragma unroll
        for (int ct = 0; ct < 2; ++ct)
#pragma unroll
            for (int dc = 0; dc < 2; ++dc) {
                // B[d, key] = K[key, d]; lane=key column, d contiguous per half
                v16bf kb = *(const v16bf*)(K + (size_t)(kt + ct * 16 + l) * HD + dc * 32 + hf * 16);
                s[ct] = wmma_bf16(qa[dc], kb, s[ct]);
            }
        // fixed-shift softmax numerator; slot i <-> row (i + hf*8), lane = key col
#pragma unroll
        for (int i = 0; i < 8; ++i) {
            const float p0 = __expf(s[0][i] - 8.0f);
            const float p1 = __expf(s[1][i] - 8.0f);
            lsum[i] += p0 + p1;                       // per-lane partial row sum
            myp[(i + hf * 8) * 32 + l]      = f2bf(p0);
            myp[(i + hf * 8) * 32 + l + 16] = f2bf(p1);
        }
        // P (16x32) back as one bf16 A-fragment
        V16U pf;
        pf.h[0] = *(const v8bf*)(myp + l * 32 + hf * 16);
        pf.h[1] = *(const v8bf*)(myp + l * 32 + hf * 16 + 8);
#pragma unroll
        for (int n = 0; n < 4; ++n) {
            // B[key, d] = V[key, d]; lane=d column -> read row of transposed V
            v16bf vb = *(const v16bf*)(Vt + (size_t)(n * 16 + l) * SEQ + kt + hf * 16);
            o[n] = wmma_bf16(pf.v, vb, o[n]);
        }
    }

#pragma unroll
    for (int i = 0; i < 8; ++i) {
        float rs = lsum[i];                           // cross-lane reduce once
        rs += __shfl_xor(rs, 8, 32);
        rs += __shfl_xor(rs, 4, 32);
        rs += __shfl_xor(rs, 2, 32);
        rs += __shfl_xor(rs, 1, 32);
        const float inv = 1.0f / rs;
        const int tok = qt + i + hf * 8;
        __bf16* po = attn + (size_t)(b * SEQ + tok) * DIM + h * HD + l;
#pragma unroll
        for (int n = 0; n < 4; ++n) po[n * 16] = f2bf(o[n][i] * inv);
    }
}

// ------------------------------------------------------------- proj GEMM
// wave tile: 32 rows x 64 cols, fp32 output + bias
__global__ void __launch_bounds__(256) k_gemm_proj(
    const __bf16* __restrict__ Xa, const __bf16* __restrict__ W,
    const float* __restrict__ bias, float* __restrict__ out)
{
    const int lane = threadIdx.x & 31;
    const int l = lane & 15, hf = lane >> 4;
    const int wave = threadIdx.x >> 5;
    const int m0 = (blockIdx.y * 8 + wave) * 32;
    const int n0 = blockIdx.x * 64;

    v8f c[2][4] = {};
    const __bf16* pa0 = Xa + (size_t)(m0 + l) * DIM + hf * 16;
    const __bf16* pa1 = Xa + (size_t)(m0 + 16 + l) * DIM + hf * 16;
    for (int k0 = 0; k0 < DIM; k0 += 32) {
        V16U a0, a1;
        a0.h[0] = *(const v8bf*)(pa0 + k0);
        a0.h[1] = *(const v8bf*)(pa0 + k0 + 8);
        a1.h[0] = *(const v8bf*)(pa1 + k0);
        a1.h[1] = *(const v8bf*)(pa1 + k0 + 8);
#pragma unroll
        for (int n = 0; n < 4; ++n) {
            v16bf b = *(const v16bf*)(W + (size_t)(n0 + n * 16 + l) * DIM + k0 + hf * 16);
            c[0][n] = wmma_bf16(a0.v, b, c[0][n]);
            c[1][n] = wmma_bf16(a1.v, b, c[1][n]);
        }
    }
#pragma unroll
    for (int n = 0; n < 4; ++n) {
        const float bv = bias[n0 + n * 16 + l];
#pragma unroll
        for (int r = 0; r < 2; ++r)
#pragma unroll
            for (int i = 0; i < 8; ++i) {
                const int t = m0 + r * 16 + i + hf * 8;
                out[(size_t)t * DIM + n0 + n * 16 + l] = c[r][n][i] + bv;
            }
    }
}

// ---------------------------------------------------------------- launcher
extern "C" void kernel_launch(void* const* d_in, const int* in_sizes, int n_in,
                              void* d_out, int out_size, void* d_ws, size_t ws_size,
                              hipStream_t stream) {
    (void)in_sizes; (void)n_in; (void)out_size; (void)ws_size;
    const float* x      = (const float*)d_in[0];
    const float* qkv_w  = (const float*)d_in[1];
    const float* qkv_b  = (const float*)d_in[2];
    const float* proj_w = (const float*)d_in[3];
    const float* proj_b = (const float*)d_in[4];
    const float* qnw    = (const float*)d_in[5];
    const float* knw    = (const float*)d_in[6];
    float* out = (float*)d_out;

    char* ws = (char*)d_ws;
    size_t off = 0;
    auto alloc = [&](size_t bytes) {
        char* p = ws + off;
        off += (bytes + 255) & ~(size_t)255;
        return p;
    };
    __bf16* x_bf     = (__bf16*)alloc((size_t)MTOT * DIM * 2);
    __bf16* qkvw_bf  = (__bf16*)alloc((size_t)DIM3 * DIM * 2);
    __bf16* projw_bf = (__bf16*)alloc((size_t)DIM  * DIM * 2);
    __bf16* q_bf     = (__bf16*)alloc((size_t)MTOT * DIM * 2);
    __bf16* k_bf     = (__bf16*)alloc((size_t)MTOT * DIM * 2);
    __bf16* v_t      = (__bf16*)alloc((size_t)MTOT * DIM * 2);
    __bf16* attn_bf  = (__bf16*)alloc((size_t)MTOT * DIM * 2);

    k_cvt<<<4096, 256, 0, stream>>>(x,      x_bf,     MTOT * DIM);
    k_cvt<<<1024, 256, 0, stream>>>(qkv_w,  qkvw_bf,  DIM3 * DIM);
    k_cvt<<<512,  256, 0, stream>>>(proj_w, projw_bf, DIM  * DIM);

    k_gemm_qkv<<<dim3(36, 32), 256, 0, stream>>>(x_bf, qkvw_bf, qkv_b, qnw, knw,
                                                 q_bf, k_bf, v_t);
    k_flash<<<1536, 128, 0, stream>>>(q_bf, k_bf, v_t, attn_bf);
    k_gemm_proj<<<dim3(12, 32), 256, 0, stream>>>(attn_bf, projw_bf, proj_b, out);
}
